// LSTMModule_57690000720396
// MI455X (gfx1250) — matrix operand order repaired and checked by reading, not compiled, not run on hardware
//
#include <hip/hip_runtime.h>
#include <hip/hip_bf16.h>

typedef unsigned short u16;
typedef __attribute__((ext_vector_type(16))) __bf16 v16bf;
typedef __attribute__((ext_vector_type(8)))  float  v8f;
typedef __attribute__((ext_vector_type(4)))  unsigned int v4u;

union Frag { v16bf bf; v4u u[2]; };

#define B_   256
#define T_   512
#define DIN  128
#define H_   256
#define OUT_ 128
#define NG   1024          // 4*H
#define K0C  384           // DIN + H   (layer 0 concat K)
#define K1C  512           // H + H     (layer 1 concat K)
#define BT   16            // batch tile (WMMA M)
#define A0S  (K0C + 8)     // padded LDS row stride (halfwords)
#define A1S  (K1C + 8)

__device__ __forceinline__ u16 f2bf(float f) {
  unsigned u = __float_as_uint(f);
  unsigned r = (u + 0x7FFFu + ((u >> 16) & 1u)) >> 16;  // round-to-nearest-even
  return (u16)r;
}

// Fast transcendentals: v_rcp_f32 + v_exp_f32 (and v_tanh_f32 when available)
__device__ __forceinline__ float fast_rcp(float x) {
  return __builtin_amdgcn_rcpf(x);                     // v_rcp_f32
}
__device__ __forceinline__ float sigm(float x) {
  return fast_rcp(1.0f + __expf(-x));                  // v_exp_f32 + v_rcp_f32
}
#if __has_builtin(__builtin_amdgcn_tanhf)
__device__ __forceinline__ float tanh_fast(float x) {
  return __builtin_amdgcn_tanhf(x);                    // v_tanh_f32
}
#elif __has_builtin(__builtin_amdgcn_tanh_f32)
__device__ __forceinline__ float tanh_fast(float x) {
  return __builtin_amdgcn_tanh_f32(x);
}
#else
__device__ __forceinline__ float tanh_fast(float x) {
  return 2.0f * sigm(2.0f * x) - 1.0f;
}
#endif

// ---------------------------------------------------------------------------
// Prep: convert weights to bf16, concat [W_ih | W_hh] row-wise, combine biases
// ---------------------------------------------------------------------------
__global__ void prep_weights(const float* __restrict__ Wih0, const float* __restrict__ Whh0,
                             const float* __restrict__ bih0, const float* __restrict__ bhh0,
                             const float* __restrict__ Wih1, const float* __restrict__ Whh1,
                             const float* __restrict__ bih1, const float* __restrict__ bhh1,
                             const float* __restrict__ Wfc,
                             u16* __restrict__ W0c, u16* __restrict__ W1c,
                             u16* __restrict__ Wfcc,
                             float* __restrict__ b0c, float* __restrict__ b1c)
{
  int i = blockIdx.x * blockDim.x + threadIdx.x;
  if (i < NG * K0C) {
    int n = i / K0C, k = i % K0C;
    float v = (k < DIN) ? Wih0[n * DIN + k] : Whh0[n * H_ + (k - DIN)];
    W0c[i] = f2bf(v);
  }
  if (i < NG * K1C) {
    int n = i / K1C, k = i % K1C;
    float v = (k < H_) ? Wih1[n * H_ + k] : Whh1[n * H_ + (k - H_)];
    W1c[i] = f2bf(v);
  }
  if (i < OUT_ * H_) Wfcc[i] = f2bf(Wfc[i]);
  if (i < NG) {
    b0c[i] = bih0[i] + bhh0[i];
    b1c[i] = bih1[i] + bhh1[i];
  }
}

// ---------------------------------------------------------------------------
// Fused persistent 2-layer LSTM + final FC.
// grid = 16 (batch tiles of 16), block = 512 (16 wave32).
// Wave w owns hidden columns [16w,16w+16) and gate tiles N = 256*g + 16*w,
// so i/f/g/o for a given hidden index share lane+slot: c stays in VGPRs.
// Weights stream from L2 (1.75 MB bf16, fully L2-resident on 192 MB L2).
// ---------------------------------------------------------------------------
__global__ __launch_bounds__(512, 1)
void lstm2_fused(const float* __restrict__ x,      // [B][T][DIN] f32
                 const u16*  __restrict__ W0,      // [NG][K0C] bf16
                 const u16*  __restrict__ W1,      // [NG][K1C] bf16
                 const u16*  __restrict__ Wfc,     // [OUT][H] bf16
                 const float* __restrict__ bias0,  // [NG]
                 const float* __restrict__ bias1,  // [NG]
                 float* __restrict__ out)          // [B][OUT] f32
{
  __shared__ u16 sA0[BT][A0S];   // [ x_t (0..127) | h0 (128..383) ]  bf16
  __shared__ u16 sA1[BT][A1S];   // [ h0  (0..255) | h1 (256..511) ]  bf16

  const int b0   = blockIdx.x * BT;
  const int tid  = threadIdx.x;
  const int wave = tid >> 5;
  const int lane = tid & 31;
  const int lh   = lane >> 4;    // lane half (A: K-subrange, D: row offset)
  const int lm   = lane & 15;

  float c0r[8], c1r[8];
#pragma unroll
  for (int r = 0; r < 8; ++r) { c0r[r] = 0.f; c1r[r] = 0.f; }

  float b0v[4], b1v[4];
#pragma unroll
  for (int g = 0; g < 4; ++g) {
    int n = g * 256 + wave * 16 + lm;
    b0v[g] = bias0[n];
    b1v[g] = bias1[n];
  }

  // zero h/c staging (h0, h1 must start at 0)
  for (int i = tid; i < BT * A0S; i += 512) (&sA0[0][0])[i] = 0;
  for (int i = tid; i < BT * A1S; i += 512) (&sA1[0][0])[i] = 0;
  __syncthreads();

#pragma unroll 1
  for (int t = 0; t < T_; ++t) {
    // stage x_t -> bf16 LDS (16 rows x 128)
    for (int i = tid; i < BT * DIN; i += 512) {
      int m = i >> 7, k = i & (DIN - 1);
      float v = x[((size_t)(b0 + m) * T_ + t) * DIN + k];
      sA0[m][k] = f2bf(v);
    }
    if (t + 1 < T_ && tid < BT) {
      __builtin_prefetch(&x[((size_t)(b0 + tid) * T_ + (t + 1)) * DIN], 0, 0);
    }
    __syncthreads();

    // ---------------- layer 0: gates = [x_t | h0] @ W0^T + b ----------------
    v8f acc[4];
#pragma unroll
    for (int g = 0; g < 4; ++g)
#pragma unroll
      for (int r = 0; r < 8; ++r) acc[g][r] = b0v[g];

#pragma unroll 2
    for (int kb = 0; kb < K0C / 32; ++kb) {
      Frag a;                                    // A: row lm, K per ISA layout
      const u16* pa = &sA0[lm][kb * 32 + lh * 8];
      a.u[0] = *(const v4u*)pa;
      a.u[1] = *(const v4u*)(pa + 16);
#pragma unroll
      for (int g = 0; g < 4; ++g) {
        int n = g * 256 + wave * 16 + lm;        // B: col n, K per ISA layout
        Frag b;
        const u16* pb = W0 + (size_t)n * K0C + kb * 32 + lh * 8;
        b.u[0] = *(const v4u*)pb;
        b.u[1] = *(const v4u*)(pb + 16);
        acc[g] = __builtin_amdgcn_wmma_f32_16x16x32_bf16(
            false, a.bf, false, b.bf, (short)0, acc[g], false, false);
      }
    }
    __syncthreads();  // all waves done reading x/h0 before h0 is overwritten

#pragma unroll
    for (int r = 0; r < 8; ++r) {
      float iv = sigm(acc[0][r]);
      float fv = sigm(acc[1][r]);
      float gv = tanh_fast(acc[2][r]);
      float ov = sigm(acc[3][r]);
      float c  = fv * c0r[r] + iv * gv;
      c0r[r] = c;
      float h = ov * tanh_fast(c);
      u16 hb = f2bf(h);
      int m = r + 8 * lh;              // D layout: lanes16-31 hold rows 8..15
      int j = wave * 16 + lm;
      sA0[m][DIN + j] = hb;            // feeds layer-0 recurrence at t+1
      sA1[m][j]       = hb;            // feeds layer-1 input at t
    }
    __syncthreads();

    // ---------------- layer 1: gates = [h0 | h1] @ W1^T + b ----------------
#pragma unroll
    for (int g = 0; g < 4; ++g)
#pragma unroll
      for (int r = 0; r < 8; ++r) acc[g][r] = b1v[g];

#pragma unroll 2
    for (int kb = 0; kb < K1C / 32; ++kb) {
      Frag a;
      const u16* pa = &sA1[lm][kb * 32 + lh * 8];
      a.u[0] = *(const v4u*)pa;
      a.u[1] = *(const v4u*)(pa + 16);
#pragma unroll
      for (int g = 0; g < 4; ++g) {
        int n = g * 256 + wave * 16 + lm;
        Frag b;
        const u16* pb = W1 + (size_t)n * K1C + kb * 32 + lh * 8;
        b.u[0] = *(const v4u*)pb;
        b.u[1] = *(const v4u*)(pb + 16);
        acc[g] = __builtin_amdgcn_wmma_f32_16x16x32_bf16(
            false, a.bf, false, b.bf, (short)0, acc[g], false, false);
      }
    }
    __syncthreads();  // all waves done reading h0/h1 before h1 is overwritten

#pragma unroll
    for (int r = 0; r < 8; ++r) {
      float iv = sigm(acc[0][r]);
      float fv = sigm(acc[1][r]);
      float gv = tanh_fast(acc[2][r]);
      float ov = sigm(acc[3][r]);
      float c  = fv * c1r[r] + iv * gv;
      c1r[r] = c;
      float h = ov * tanh_fast(c);
      int m = r + 8 * lh;
      int j = wave * 16 + lm;
      sA1[m][H_ + j] = f2bf(h);
    }
    __syncthreads();
  }

  // ---------------- final FC: out = h1[T-1] @ Wfc^T (16x128x256) ----------
  if (wave < 8) {                      // 8 N-tiles of 16; branch wave-uniform
    v8f acc;
#pragma unroll
    for (int r = 0; r < 8; ++r) acc[r] = 0.f;
#pragma unroll
    for (int kb = 0; kb < H_ / 32; ++kb) {
      Frag a;
      const u16* pa = &sA1[lm][H_ + kb * 32 + lh * 8];
      a.u[0] = *(const v4u*)pa;
      a.u[1] = *(const v4u*)(pa + 16);
      Frag b;
      int n = wave * 16 + lm;
      const u16* pb = Wfc + (size_t)n * H_ + kb * 32 + lh * 8;
      b.u[0] = *(const v4u*)pb;
      b.u[1] = *(const v4u*)(pb + 16);
      acc = __builtin_amdgcn_wmma_f32_16x16x32_bf16(
          false, a.bf, false, b.bf, (short)0, acc, false, false);
    }
#pragma unroll
    for (int r = 0; r < 8; ++r) {
      int m = r + 8 * lh;
      out[(size_t)(b0 + m) * OUT_ + wave * 16 + lm] = acc[r];
    }
  }
}

// ---------------------------------------------------------------------------
extern "C" void kernel_launch(void* const* d_in, const int* in_sizes, int n_in,
                              void* d_out, int out_size, void* d_ws, size_t ws_size,
                              hipStream_t stream)
{
  const float* x    = (const float*)d_in[0];
  const float* Wih0 = (const float*)d_in[1];
  const float* Whh0 = (const float*)d_in[2];
  const float* bih0 = (const float*)d_in[3];
  const float* bhh0 = (const float*)d_in[4];
  const float* Wih1 = (const float*)d_in[5];
  const float* Whh1 = (const float*)d_in[6];
  const float* bih1 = (const float*)d_in[7];
  const float* bhh1 = (const float*)d_in[8];
  const float* Wfc  = (const float*)d_in[9];

  // workspace layout (bytes), ~1.9 MB total
  char* ws = (char*)d_ws;
  u16*   W0c  = (u16*)(ws + 0);              // 1024*384*2 = 786432
  u16*   W1c  = (u16*)(ws + 786432);         // 1024*512*2 = 1048576
  u16*   Wfcc = (u16*)(ws + 1835008);        // 128*256*2  = 65536
  float* b0c  = (float*)(ws + 1900544);      // 4096
  float* b1c  = (float*)(ws + 1904640);      // 4096

  int total = NG * K1C;  // largest array: 524288 elems covers everything
  prep_weights<<<(total + 255) / 256, 256, 0, stream>>>(
      Wih0, Whh0, bih0, bhh0, Wih1, Whh1, bih1, bhh1, Wfc,
      W0c, W1c, Wfcc, b0c, b1c);

  lstm2_fused<<<B_ / BT, 512, 0, stream>>>(
      x, W0c, W1c, Wfcc, b0c, b1c, (float*)d_out);
}
